// IterativeCancellationBlock_76579266888370
// MI455X (gfx1250) — matrix operand order repaired and checked by reading, not compiled, not run on hardware
//
#include <hip/hip_runtime.h>
#include <hip/hip_bf16.h>
#include <math.h>

// Problem constants (match reference)
constexpr int kB = 2;
constexpr int kL = 2048;
constexpr int kD = 256;
constexpr int kP = 32;
constexpr int kH = 512;          // 2*D
constexpr int kF = 64;           // feature dim = 2*P (cos,sin)
constexpr int kC = 64;           // chunk (tokens per workgroup)
constexpr int kNC = kL / kC;     // 32 chunks per batch
constexpr int kNBLK = kB * kL / kC;  // 64 blocks total

typedef _Float16 h8v  __attribute__((ext_vector_type(8)));
typedef _Float16 h16v __attribute__((ext_vector_type(16)));
typedef float    f8v  __attribute__((ext_vector_type(8)));
typedef unsigned int u32x4 __attribute__((ext_vector_type(4)));
typedef int          i32x4 __attribute__((ext_vector_type(4)));
typedef int          i32x8 __attribute__((ext_vector_type(8)));

// ---------------- WMMA fragment helpers (wave32, 16x16x32 f16) ----------------
// A fragment: A[m][k], source row-major [M][K] with leading dim ld (halves).
//   lanes 0-15: row m=lane, K in {k0..k0+7} (v0-3) and {k0+16..k0+23} (v4-7)
//   lanes16-31: row m=lane-16, K in {k0+8..k0+15} and {k0+24..k0+31}
__device__ __forceinline__ h16v load_frag_a(const _Float16* __restrict__ M,
                                            int ld, int row0, int k0) {
  int lane = threadIdx.x & 31;
  int r  = row0 + (lane & 15);
  int kb = k0 + ((lane >> 4) << 3);
  const _Float16* p = M + (size_t)r * ld + kb;
  h8v lo = *(const h8v*)p;
  h8v hi = *(const h8v*)(p + 16);
  return __builtin_shufflevector(lo, hi, 0,1,2,3,4,5,6,7,8,9,10,11,12,13,14,15);
}

// B fragment: B[k][n], source stored TRANSPOSED row-major as BT[N][K], ld in halves.
//   lanes 0-15: col n=lane, K in {k0..k0+7} and {k0+16..k0+23}; lanes16-31: K in {k0+8..k0+15} and {k0+24..k0+31}
__device__ __forceinline__ h16v load_frag_b(const _Float16* __restrict__ BT,
                                            int ld, int col0, int k0) {
  int lane = threadIdx.x & 31;
  int r  = col0 + (lane & 15);
  int kb = k0 + ((lane >> 4) << 3);
  const _Float16* p = BT + (size_t)r * ld + kb;
  h8v lo = *(const h8v*)p;
  h8v hi = *(const h8v*)(p + 16);
  return __builtin_shufflevector(lo, hi, 0,1,2,3,4,5,6,7,8,9,10,11,12,13,14,15);
}

__device__ __forceinline__ f8v wmma_f16(h16v a, h16v b, f8v c) {
  return __builtin_amdgcn_wmma_f32_16x16x32_f16(false, a, false, b,
                                                (short)0, c, false, false);
}

__device__ __forceinline__ f8v zero8() {
  f8v c = {0.f,0.f,0.f,0.f,0.f,0.f,0.f,0.f};
  return c;
}

// ---------------- TDM: 2D f16 tile load (tile == tensor), per ISA 08 D# layout
#if __has_builtin(__builtin_amdgcn_tensor_load_to_lds)
__device__ __forceinline__ void tdm_load_2d_f16(unsigned lds_addr, const void* gptr,
                                                unsigned dim0, unsigned dim1) {
  unsigned long long ga = (unsigned long long)(uintptr_t)gptr;
  u32x4 g0;
  g0[0] = 1u;                                        // count=1, no gather
  g0[1] = lds_addr;                                  // lds_addr [63:32]
  g0[2] = (unsigned)ga;                              // global_addr [95:64]
  g0[3] = (unsigned)((ga >> 32) & 0x01FFFFFFu) | 0x80000000u;  // addr hi | type=2
  i32x8 g1;
  g1[0] = 0x00010000;                                // data_size=1 (2 bytes)
  g1[1] = (int)(dim0 << 16);                         // tensor_dim0 lo16 -> [63:48]
  g1[2] = (int)((dim0 >> 16) | (dim1 << 16));        // dim0 hi | tensor_dim1 lo
  g1[3] = (int)((dim1 >> 16) | (dim0 << 16));        // dim1 hi | tile_dim0
  g1[4] = (int)dim1;                                 // tile_dim1 (tile_dim2=0)
  g1[5] = (int)dim0;                                 // tensor_dim0_stride lo32
  g1[6] = 0;
  g1[7] = 0;                                         // tensor_dim1_stride unused (2D)
  i32x4 gz;
  gz[0] = 0; gz[1] = 0; gz[2] = 0; gz[3] = 0;
  i32x8 gz8;
  gz8[0] = 0; gz8[1] = 0; gz8[2] = 0; gz8[3] = 0;
  gz8[4] = 0; gz8[5] = 0; gz8[6] = 0; gz8[7] = 0;
  // amdgpu-toolchain (clang-23) 6-arg form
  __builtin_amdgcn_tensor_load_to_lds(g0, g1, gz, gz, gz8, 0);
}
#endif

// ---------------- weight convert + transpose: src fp32 [R][C] -> dst f16 [C][R]
__global__ void k_convT(const float* __restrict__ src, _Float16* __restrict__ dst,
                        int R, int C) {
  int i = blockIdx.x * blockDim.x + threadIdx.x;
  if (i >= R * C) return;
  int r = i / C, c = i - r * C;
  dst[(size_t)c * R + r] = (_Float16)src[i];
}

// ---------------- featurize: phases -> F=[cos,sin] (f16, row + chunk-transposed),
//                  values -> u (fp32 row-major) and u16T (f16 [blk][d][tok])
__global__ void k_featurize(const float* __restrict__ x,
                            const float* __restrict__ pe_b,
                            const float* __restrict__ tv_b,
                            const _Float16* __restrict__ peT,   // [32][256]
                            const _Float16* __restrict__ tvT,   // [256][256]
                            _Float16* __restrict__ Fmat,        // [B*L][64]
                            _Float16* __restrict__ FT,          // [blk][64][64]
                            float* __restrict__ u,              // [B*L][256]
                            _Float16* __restrict__ u16T) {      // [blk][256][64]
  __shared__ _Float16 Xl[64][264];
  int blk = blockIdx.x, tid = threadIdx.x, wave = tid >> 5, lane = tid & 31;
  const float* xblk = x + (size_t)blk * kC * kD;
  for (int i = tid; i < kC * kD; i += 256)
    Xl[i >> 8][i & 255] = (_Float16)xblk[i];
  __syncthreads();

  int n_l = lane & 15, mbase = (lane >> 4) << 3;

  // phases GEMM: [64,256] x [256,32] -> 8 tiles (one per wave)
  {
    int tm = wave & 3, tn = wave >> 2;   // tn in 0..1
    f8v c = zero8();
    for (int k = 0; k < kD; k += 32) {
      h16v a = load_frag_a(&Xl[0][0], 264, tm * 16, k);
      h16v b = load_frag_b(peT, kD, tn * 16, k);
      c = wmma_f16(a, b, c);
    }
    int n = tn * 16 + n_l;
    float bias = pe_b[n];
    for (int r = 0; r < 8; ++r) {
      int m = tm * 16 + r + mbase;
      float ph = tanhf(c[r] + bias) * 3.14159265358979323846f;
      float cs = __builtin_cosf(ph), sn = __builtin_sinf(ph);
      size_t tok = (size_t)blk * kC + m;
      Fmat[tok * kF + n]      = (_Float16)cs;
      Fmat[tok * kF + kP + n] = (_Float16)sn;
      _Float16* ft = FT + (size_t)blk * kC * kF;
      ft[(size_t)n * kC + m]        = (_Float16)cs;
      ft[(size_t)(kP + n) * kC + m] = (_Float16)sn;
    }
  }

  // values GEMM: [64,256] x [256,256] -> 64 tiles, 8 per wave
  for (int t = wave; t < 64; t += 8) {
    int tm = t & 3, tn = t >> 2;
    f8v c = zero8();
    for (int k = 0; k < kD; k += 32) {
      h16v a = load_frag_a(&Xl[0][0], 264, tm * 16, k);
      h16v b = load_frag_b(tvT, kD, tn * 16, k);
      c = wmma_f16(a, b, c);
    }
    int n = tn * 16 + n_l;
    float bias = tv_b[n];
    for (int r = 0; r < 8; ++r) {
      int m = tm * 16 + r + mbase;
      float v = c[r] + bias;
      u[(size_t)(blk * kC + m) * kD + n] = v;
      u16T[(size_t)blk * kD * kC + (size_t)n * kC + m] = (_Float16)v;
    }
  }
}

// ---------------- per-chunk K^T V:  ktv[blk][p2][d] = sum_tok F[tok][p2]*u[tok][d]
// Operand tiles staged into LDS via the Tensor Data Mover (one wave issues the
// DMA descriptors, TENSORcnt-tracked), then all WMMA fragments come from LDS.
__global__ void k_ktv(const _Float16* __restrict__ FT,     // [blk][p2][tok]
                      const _Float16* __restrict__ u16T,   // [blk][d][tok]
                      float* __restrict__ ktv) {           // [blk][64][256]
  __shared__ _Float16 FTl[64][64];    // 8 KB
  __shared__ _Float16 UTl[256][64];   // 32 KB
  int blk = blockIdx.x, tid = threadIdx.x, wave = tid >> 5, lane = tid & 31;
  const _Float16* Fg = FT   + (size_t)blk * kF * kC;
  const _Float16* Ug = u16T + (size_t)blk * kD * kC;
#if __has_builtin(__builtin_amdgcn_tensor_load_to_lds)
  if (wave == 0) {
    tdm_load_2d_f16((unsigned)(uintptr_t)&FTl[0][0], Fg, kC, kF);
    tdm_load_2d_f16((unsigned)(uintptr_t)&UTl[0][0], Ug, kC, kD);
    __builtin_amdgcn_s_wait_tensorcnt(0);
  }
#else
  for (int i = tid; i < kF * kC; i += 256) FTl[i >> 6][i & 63] = Fg[i];
  for (int i = tid; i < kD * kC; i += 256) UTl[i >> 6][i & 63] = Ug[i];
#endif
  __syncthreads();

  float* out = ktv + (size_t)blk * kF * kD;
  int n_l = lane & 15, mbase = (lane >> 4) << 3;
  for (int t = wave; t < 64; t += 8) {
    int tm = t & 3, tn = t >> 2;
    f8v c = zero8();
    for (int k = 0; k < kC; k += 32) {
      h16v a = load_frag_a(&FTl[0][0], kC, tm * 16, k);
      h16v b = load_frag_b(&UTl[0][0], kC, tn * 16, k);
      c = wmma_f16(a, b, c);
    }
    int n = tn * 16 + n_l;
    for (int r = 0; r < 8; ++r)
      out[(size_t)(tm * 16 + r + mbase) * kD + n] = c[r];
  }
}

// ---------------- exclusive scan over chunks; emit state transposed f16 [d][p2]
__global__ void k_scan(const float* __restrict__ ktv, _Float16* __restrict__ spreT) {
  int i = blockIdx.x * blockDim.x + threadIdx.x;     // b*16384 + (p*256 + d)
  int b = i >> 14, e = i & 16383;
  int p = e >> 8, d = e & 255;
  const float* src = ktv + ((size_t)b * kNC << 14) + e;
  _Float16* dst = spreT + ((size_t)b * kNC << 14) + (size_t)d * kC + p;
  float run = 0.f;
  for (int c = 0; c < kNC; ++c) {
    float v = src[(size_t)c << 14];
    dst[(size_t)c << 14] = (_Float16)run;
    run += v;
  }
}

// ---------------- attn output: retrieved = (mask(F F^T) U + F S_prefix) / norm
__global__ void k_attn_out(const _Float16* __restrict__ Fmat,  // [B*L][64]
                           const _Float16* __restrict__ u16T,  // [blk][d][tok]
                           const _Float16* __restrict__ spreT, // [blk][d][p2]
                           float* __restrict__ retr) {         // [B*L][256]
  __shared__ _Float16 As[64][72];
  int blk = blockIdx.x, chunk = blk & (kNC - 1);
  int tid = threadIdx.x, wave = tid >> 5, lane = tid & 31;
  int n_l = lane & 15, mbase = (lane >> 4) << 3;
  const _Float16* Fc = Fmat  + (size_t)blk * kC * kF;
  const _Float16* Uc = u16T  + (size_t)blk * kD * kC;
  const _Float16* Sc = spreT + (size_t)blk * kD * kC;
  float* out = retr + (size_t)blk * kC * kD;

  // phase 1: scores = F F^T with causal mask (inclusive diagonal)
  for (int t = wave; t < 16; t += 8) {
    int tm = t & 3, tn = t >> 2;
    f8v c = zero8();
    for (int k = 0; k < kF; k += 32) {
      h16v a = load_frag_a(Fc, kF, tm * 16, k);
      h16v b = load_frag_b(Fc, kF, tn * 16, k);  // F rows double as BT[N=tok'][K=p2]
      c = wmma_f16(a, b, c);
    }
    int n = tn * 16 + n_l;
    for (int r = 0; r < 8; ++r) {
      int m = tm * 16 + r + mbase;
      As[m][n] = (n <= m) ? (_Float16)c[r] : (_Float16)0.0f;
    }
  }
  __syncthreads();

  // phase 2: As @ U + F @ S, scaled by 1/(sqrt(l+1)*sqrt(P))
  for (int t = wave; t < 64; t += 8) {
    int tm = t & 3, tn = t >> 2;
    f8v c = zero8();
    for (int k = 0; k < kC; k += 32) {
      h16v a  = load_frag_a(&As[0][0], 72, tm * 16, k);
      h16v b  = load_frag_b(Uc, kC, tn * 16, k);
      c = wmma_f16(a, b, c);
      h16v a2 = load_frag_a(Fc, kF, tm * 16, k);
      h16v b2 = load_frag_b(Sc, kC, tn * 16, k);
      c = wmma_f16(a2, b2, c);
    }
    int n = tn * 16 + n_l;
    for (int r = 0; r < 8; ++r) {
      int m = tm * 16 + r + mbase;
      int l = chunk * kC + m;
      out[(size_t)m * kD + n] = c[r] * rsqrtf((float)(l + 1) * (float)kP);
    }
  }
}

// ---------------- cleanup MLP: LN -> GEMM(D->H) -> GELU -> GEMM(H->D)
__global__ void k_cleanup(const float* __restrict__ retr,
                          const float* __restrict__ ln_g, const float* __restrict__ ln_b,
                          const _Float16* __restrict__ w1T,  // [H][D]
                          const float* __restrict__ b1,
                          const _Float16* __restrict__ w2T,  // [D][H]
                          const float* __restrict__ b2,
                          const float* __restrict__ cancel_w, int iter, int update_u,
                          float* __restrict__ acc, float* __restrict__ u,
                          _Float16* __restrict__ u16T,
                          _Float16* __restrict__ g16) {      // [B*L][512] scratch
  __shared__ _Float16 Hl[64][264];
  int blk = blockIdx.x, tid = threadIdx.x, wave = tid >> 5, lane = tid & 31;
  const float* Rc = retr + (size_t)blk * kC * kD;

  // LayerNorm: each wave normalizes 8 rows via shfl reductions
  for (int rr = 0; rr < 8; ++rr) {
    int row = wave * 8 + rr;
    const float* rp = Rc + (size_t)row * kD;
    float vals[8], s = 0.f;
    for (int j = 0; j < 8; ++j) { vals[j] = rp[lane * 8 + j]; s += vals[j]; }
    for (int off = 16; off > 0; off >>= 1) s += __shfl_xor(s, off, 32);
    float mu = s * (1.0f / 256.0f);
    float vs = 0.f;
    for (int j = 0; j < 8; ++j) { float d0 = vals[j] - mu; vs += d0 * d0; }
    for (int off = 16; off > 0; off >>= 1) vs += __shfl_xor(vs, off, 32);
    float inv = rsqrtf(vs * (1.0f / 256.0f) + 1e-5f);
    for (int j = 0; j < 8; ++j) {
      int d = lane * 8 + j;
      Hl[row][d] = (_Float16)((vals[j] - mu) * inv * ln_g[d] + ln_b[d]);
    }
  }
  __syncthreads();

  int n_l = lane & 15, mbase = (lane >> 4) << 3;
  _Float16* gblk = g16 + (size_t)blk * kC * kH;

  // GEMM1 + exact GELU: [64,256] x [256,512], 128 tiles
  for (int t = wave; t < 128; t += 8) {
    int tm = t & 3, tn = t >> 2;
    f8v c = zero8();
    for (int k = 0; k < kD; k += 32) {
      __builtin_prefetch(w1T + (size_t)(tn * 16) * kD + k + 32, 0, 1);
      h16v a = load_frag_a(&Hl[0][0], 264, tm * 16, k);
      h16v b = load_frag_b(w1T, kD, tn * 16, k);
      c = wmma_f16(a, b, c);
    }
    int n = tn * 16 + n_l;
    float bias = b1[n];
    for (int r = 0; r < 8; ++r) {
      int m = tm * 16 + r + mbase;
      float hv = c[r] + bias;
      float g = 0.5f * hv * (1.0f + erff(hv * 0.70710678118654752f));
      gblk[(size_t)m * kH + n] = (_Float16)g;
    }
  }
  __syncthreads();

  float cw = 0.f;
  if (update_u) cw = 1.0f / (1.0f + expf(-cancel_w[iter]));

  // GEMM2: [64,512] x [512,256], 64 tiles
  for (int t = wave; t < 64; t += 8) {
    int tm = t & 3, tn = t >> 2;
    f8v c = zero8();
    for (int k = 0; k < kH; k += 32) {
      h16v a = load_frag_a(gblk, kH, tm * 16, k);
      h16v b = load_frag_b(w2T, kH, tn * 16, k);
      c = wmma_f16(a, b, c);
    }
    int n = tn * 16 + n_l;
    float bias = b2[n];
    for (int r = 0; r < 8; ++r) {
      int m = tm * 16 + r + mbase;
      size_t idx = (size_t)(blk * kC + m) * kD + n;
      float cl = c[r] + bias;
      acc[idx] += cl;
      if (update_u) {
        float nu = u[idx] - cw * cl;
        u[idx] = nu;
        u16T[(size_t)blk * kD * kC + (size_t)n * kC + m] = (_Float16)nu;
      }
    }
  }
}

// ---------------- final: out = x + LN(acc) @ out_w + out_b
__global__ void k_final(const float* __restrict__ acc,
                        const float* __restrict__ g, const float* __restrict__ bt,
                        const _Float16* __restrict__ outT,  // [256][256]
                        const float* __restrict__ out_b,
                        const float* __restrict__ x, float* __restrict__ out) {
  __shared__ _Float16 Hl[64][264];
  int blk = blockIdx.x, tid = threadIdx.x, wave = tid >> 5, lane = tid & 31;
  const float* Ac = acc + (size_t)blk * kC * kD;
  for (int rr = 0; rr < 8; ++rr) {
    int row = wave * 8 + rr;
    const float* rp = Ac + (size_t)row * kD;
    float vals[8], s = 0.f;
    for (int j = 0; j < 8; ++j) { vals[j] = rp[lane * 8 + j]; s += vals[j]; }
    for (int off = 16; off > 0; off >>= 1) s += __shfl_xor(s, off, 32);
    float mu = s * (1.0f / 256.0f);
    float vs = 0.f;
    for (int j = 0; j < 8; ++j) { float d0 = vals[j] - mu; vs += d0 * d0; }
    for (int off = 16; off > 0; off >>= 1) vs += __shfl_xor(vs, off, 32);
    float inv = rsqrtf(vs * (1.0f / 256.0f) + 1e-5f);
    for (int j = 0; j < 8; ++j) {
      int d = lane * 8 + j;
      Hl[row][d] = (_Float16)((vals[j] - mu) * inv * g[d] + bt[d]);
    }
  }
  __syncthreads();
  int n_l = lane & 15, mbase = (lane >> 4) << 3;
  for (int t = wave; t < 64; t += 8) {
    int tm = t & 3, tn = t >> 2;
    f8v c = zero8();
    for (int k = 0; k < kD; k += 32) {
      h16v a = load_frag_a(&Hl[0][0], 264, tm * 16, k);
      h16v b = load_frag_b(outT, kD, tn * 16, k);
      c = wmma_f16(a, b, c);
    }
    int n = tn * 16 + n_l;
    float bias = out_b[n];
    for (int r = 0; r < 8; ++r) {
      int m = tm * 16 + r + mbase;
      size_t idx = (size_t)(blk * kC + m) * kD + n;
      out[idx] = x[idx] + c[r] + bias;
    }
  }
}

// ---------------- host ----------------
extern "C" void kernel_launch(void* const* d_in, const int* in_sizes, int n_in,
                              void* d_out, int out_size, void* d_ws, size_t ws_size,
                              hipStream_t stream) {
  const float* x        = (const float*)d_in[0];
  const float* pe_w     = (const float*)d_in[1];
  const float* pe_b     = (const float*)d_in[2];
  const float* tv_w     = (const float*)d_in[3];
  const float* tv_b     = (const float*)d_in[4];
  const float* cu_ln_g  = (const float*)d_in[5];
  const float* cu_ln_b  = (const float*)d_in[6];
  const float* cu_w1    = (const float*)d_in[7];
  const float* cu_b1    = (const float*)d_in[8];
  const float* cu_w2    = (const float*)d_in[9];
  const float* cu_b2    = (const float*)d_in[10];
  const float* cancel_w = (const float*)d_in[11];
  const float* out_ln_g = (const float*)d_in[12];
  const float* out_ln_b = (const float*)d_in[13];
  const float* out_w    = (const float*)d_in[14];
  const float* out_b    = (const float*)d_in[15];
  float* out = (float*)d_out;

  char* ws = (char*)d_ws;
  size_t off = 0;
  auto alloc = [&](size_t bytes) -> void* {
    void* p = ws + off;
    off = (off + bytes + 255) & ~(size_t)255;
    return p;
  };

  _Float16* peT   = (_Float16*)alloc((size_t)kP * kD * 2);
  _Float16* tvT   = (_Float16*)alloc((size_t)kD * kD * 2);
  _Float16* w1T   = (_Float16*)alloc((size_t)3 * kH * kD * 2);
  _Float16* w2T   = (_Float16*)alloc((size_t)3 * kD * kH * 2);
  _Float16* outT  = (_Float16*)alloc((size_t)kD * kD * 2);
  _Float16* Fmat  = (_Float16*)alloc((size_t)kB * kL * kF * 2);
  _Float16* FT    = (_Float16*)alloc((size_t)kB * kL * kF * 2);
  float*    u     = (float*)   alloc((size_t)kB * kL * kD * 4);
  _Float16* u16T  = (_Float16*)alloc((size_t)kB * kL * kD * 2);
  _Float16* g16   = (_Float16*)alloc((size_t)kB * kL * kH * 2);
  float*    ktv   = (float*)   alloc((size_t)kB * kNC * kF * kD * 4);
  _Float16* spreT = (_Float16*)alloc((size_t)kB * kNC * kF * kD * 2);
  float*    retr  = (float*)   alloc((size_t)kB * kL * kD * 4);
  float*    accb  = (float*)   alloc((size_t)kB * kL * kD * 4);

  // weight convert+transpose
  k_convT<<<(kD * kP + 255) / 256, 256, 0, stream>>>(pe_w, peT, kD, kP);
  k_convT<<<(kD * kD + 255) / 256, 256, 0, stream>>>(tv_w, tvT, kD, kD);
  for (int i = 0; i < 3; ++i) {
    k_convT<<<(kD * kH + 255) / 256, 256, 0, stream>>>(cu_w1 + (size_t)i * kD * kH,
                                                       w1T + (size_t)i * kH * kD, kD, kH);
    k_convT<<<(kD * kH + 255) / 256, 256, 0, stream>>>(cu_w2 + (size_t)i * kH * kD,
                                                       w2T + (size_t)i * kD * kH, kH, kD);
  }
  k_convT<<<(kD * kD + 255) / 256, 256, 0, stream>>>(out_w, outT, kD, kD);

  (void)hipMemsetAsync(accb, 0, (size_t)kB * kL * kD * 4, stream);

  k_featurize<<<kNBLK, 256, 0, stream>>>(x, pe_b, tv_b, peT, tvT, Fmat, FT, u, u16T);

  for (int pass = 0; pass < 4; ++pass) {
    k_ktv<<<kNBLK, 256, 0, stream>>>(FT, u16T, ktv);
    k_scan<<<(kB * kF * kD) / 256, 256, 0, stream>>>(ktv, spreT);
    k_attn_out<<<kNBLK, 256, 0, stream>>>(Fmat, u16T, spreT, retr);
    int it  = pass < 3 ? pass : 2;
    int upd = pass < 3 ? 1 : 0;
    k_cleanup<<<kNBLK, 256, 0, stream>>>(retr, cu_ln_g + (size_t)it * kD,
                                         cu_ln_b + (size_t)it * kD,
                                         w1T + (size_t)it * kH * kD, cu_b1 + (size_t)it * kH,
                                         w2T + (size_t)it * kD * kH, cu_b2 + (size_t)it * kD,
                                         cancel_w, it, upd, accb, u, u16T, g16);
  }

  k_final<<<kNBLK, 256, 0, stream>>>(accb, out_ln_g, out_ln_b, outT, out_b, x, out);
}
